// Generator_36378372997646
// MI455X (gfx1250) — matrix operand order repaired and checked by reading, not compiled, not run on hardware
//
#include <hip/hip_runtime.h>
#include <hip/hip_bf16.h>
#include <math.h>

typedef __attribute__((ext_vector_type(16))) _Float16 v16h;
typedef __attribute__((ext_vector_type(8)))  float    v8f;

#define NROWS   2048      // B*N
#define NDIM    64        // NOISE_DIM
#define HID     128
#define NF      128
#define LDH     136       // padded f16 LDS stride (272B = 17*16 -> conflict-free A-frag reads)

// ---------- WMMA fragment helpers (layouts per CDNA5 ISA 7.12.2, wave32) ----------

__device__ __forceinline__ v8f wmma_f16(v16h a, v16h b, v8f c) {
  // D(f32 16x16) = A(f16 16x32) * B(f16 32x16) + C
  return __builtin_amdgcn_wmma_f32_16x16x32_f16(false, a, false, b, (short)0, c, false, false);
}

// A-frag (16x32, f16) from row-major f32 src (top-left already offset), leading dim lda.
// lane l holds row m=l&15; element 2v+s is K = 2v+s + (v>=4?8:0) + (l>=16?8:0).
__device__ __forceinline__ v16h load_a_frag_f32(const float* __restrict__ src, int lda, int lane) {
  int m  = lane & 15;
  int kb = (lane >= 16) ? 8 : 0;
  v16h a;
#pragma unroll
  for (int v = 0; v < 8; ++v) {
    int k = 2 * v + (v >= 4 ? 8 : 0) + kb;
    const float* p = src + m * lda + k;
    a[2 * v]     = (_Float16)p[0];
    a[2 * v + 1] = (_Float16)p[1];
  }
  return a;
}

// A-frag from row-major f16 LDS tile, leading dim ld (in elements).
__device__ __forceinline__ v16h load_a_frag_f16(const _Float16* src, int ld, int lane) {
  int m  = lane & 15;
  int kb = (lane >= 16) ? 8 : 0;
  v16h a;
#pragma unroll
  for (int v = 0; v < 8; ++v) {
    int k = 2 * v + (v >= 4 ? 8 : 0) + kb;
    a[2 * v]     = src[m * ld + k];
    a[2 * v + 1] = src[m * ld + k + 1];
  }
  return a;
}

// B-frag (32x16, f16) from row-major f32 weights (top-left already offset), leading dim ldb.
// lane l holds col n=l&15; element 2v+s is K = 2v+s + (v>=4?8:0) + (l>=16?8:0).
__device__ __forceinline__ v16h load_b_frag_f32(const float* __restrict__ src, int ldb, int lane) {
  int n  = lane & 15;
  int kb = (lane >= 16) ? 8 : 0;
  v16h b;
#pragma unroll
  for (int v = 0; v < 8; ++v) {
    int k = kb + 2 * v + (v >= 4 ? 8 : 0);
    b[2 * v]     = (_Float16)src[k * ldb + n];
    b[2 * v + 1] = (_Float16)src[(k + 1) * ldb + n];
  }
  return b;
}

// ---------- Fused node-MLP + edge-projection kernel ----------
// Each block: 16 rows of the 2048-row batch. 8 waves; wave w owns N-tile w (cols 16w..16w+15).
__global__ __launch_bounds__(256) void gen_feats_kernel(
    const float* __restrict__ Z,    // 2048 x 64
    const float* __restrict__ W1,   // 64  x 128
    const float* __restrict__ b1,   // 128
    const float* __restrict__ W2,   // 128 x 128
    const float* __restrict__ b2,   // 128
    const float* __restrict__ eW1,  // 256 x 128  (Wa = rows 0..127, Wb = rows 128..255)
    const float* __restrict__ eb1,  // 128
    float* __restrict__ feats,      // 2048 x 128 (d_out, fp32)
    float* __restrict__ HA,         // 2048 x 128 ws: ha + eb1
    float* __restrict__ HB)         // 2048 x 128 ws: hb
{
  __shared__ _Float16 h_lds[16 * LDH];   // relu(Z@W1+b1) in f16
  __shared__ _Float16 f_lds[16 * LDH];   // feats in f16

  const int lane = threadIdx.x & 31;
  const int wid  = threadIdx.x >> 5;     // N-tile 0..7
  const int row0 = blockIdx.x * 16;      // global row base
  const int n    = lane & 15;
  const int mb   = (lane >= 16) ? 8 : 0; // C-frag row base per lane half

  // ---- Layer 1: H = relu(Z @ W1 + b1) ----
  v8f c = {};
#pragma unroll
  for (int k0 = 0; k0 < NDIM; k0 += 32) {
    v16h a = load_a_frag_f32(Z + row0 * NDIM + k0, NDIM, lane);
    v16h b = load_b_frag_f32(W1 + k0 * HID + wid * 16, HID, lane);
    c = wmma_f16(a, b, c);
  }
  {
    float bias = b1[wid * 16 + n];
#pragma unroll
    for (int r = 0; r < 8; ++r) {
      float v = c[r] + bias;
      v = v > 0.0f ? v : 0.0f;
      h_lds[(mb + r) * LDH + wid * 16 + n] = (_Float16)v;
    }
  }
  __syncthreads();

  // ---- Layer 2: feats = H @ W2 + b2 ----
  c = v8f{};
#pragma unroll
  for (int k0 = 0; k0 < HID; k0 += 32) {
    v16h a = load_a_frag_f16(h_lds + k0, LDH, lane);
    v16h b = load_b_frag_f32(W2 + k0 * NF + wid * 16, NF, lane);
    c = wmma_f16(a, b, c);
  }
  {
    float bias = b2[wid * 16 + n];
#pragma unroll
    for (int r = 0; r < 8; ++r) {
      float v = c[r] + bias;
      feats[(size_t)(row0 + mb + r) * NF + wid * 16 + n] = v;
      f_lds[(mb + r) * LDH + wid * 16 + n] = (_Float16)v;
    }
  }
  __syncthreads();

  // ---- ha = feats @ Wa + eb1 ; hb = feats @ Wb ----
  v8f ca = {};
  v8f cb = {};
#pragma unroll
  for (int k0 = 0; k0 < NF; k0 += 32) {
    v16h a  = load_a_frag_f16(f_lds + k0, LDH, lane);
    v16h ba = load_b_frag_f32(eW1 + k0 * HID + wid * 16, HID, lane);
    v16h bb = load_b_frag_f32(eW1 + (NF + k0) * HID + wid * 16, HID, lane);
    ca = wmma_f16(a, ba, ca);
    cb = wmma_f16(a, bb, cb);
  }
  {
    float bias = eb1[wid * 16 + n];
#pragma unroll
    for (int r = 0; r < 8; ++r) {
      size_t idx = (size_t)(row0 + mb + r) * HID + wid * 16 + n;
      HA[idx] = ca[r] + bias;   // eb1 folded into ha
      HB[idx] = cb[r];
    }
  }
}

// Async global->LDS 16-byte copy (CDNA5 GLOBAL_LOAD_ASYNC_TO_LDS_B128, ASYNCcnt).
// ldsp must be a pointer into __shared__; low 32 bits of the generic address are
// the LDS byte offset.
__device__ __forceinline__ void async_copy_b128(void* ldsp, const void* gp) {
  uint32_t lds = (uint32_t)(size_t)ldsp;
  uint64_t ga  = (uint64_t)(size_t)gp;
  asm volatile("global_load_async_to_lds_b128 %0, %1, off"
               :: "v"(lds), "v"(ga) : "memory");
}

// ---------- Edge scorer: adj = 0.5*(sigmoid(L) + sigmoid(L)^T) ----------
// L[i,j] = sum_h relu(HA[i,h] + HB[j,h]) * w[h] + eb2.
// Grid (64,64,2); block (I,J,b) with J>=I computes both orientations of its 16x16 tile.
__global__ __launch_bounds__(256) void edge_kernel(
    const float* __restrict__ HA,   // 2048 x 128 (bias folded)
    const float* __restrict__ HB,   // 2048 x 128
    const float* __restrict__ w,    // eW2: 128
    const float* __restrict__ eb2,  // 1
    float* __restrict__ adj)        // 2 x 1024 x 1024
{
  const int I = blockIdx.x, J = blockIdx.y, b = blockIdx.z;
  if (J < I) return;

  __shared__ __align__(16) float haI[16][132];
  __shared__ __align__(16) float hbJ[16][132];
  __shared__ __align__(16) float haJ[16][132];
  __shared__ __align__(16) float hbI[16][132];
  __shared__ __align__(16) float wv[128];

  const int t = threadIdx.x;
  const int rbase = b * 1024;

  // Stage 4 tiles (16 x 128 f32 each) with async global->LDS b128 DMA:
  // 512 xfers per tile, 2 per thread per tile.
  {
    const float* sAI = HA + (size_t)(rbase + I * 16) * 128;
    const float* sAJ = HA + (size_t)(rbase + J * 16) * 128;
    const float* sBI = HB + (size_t)(rbase + I * 16) * 128;
    const float* sBJ = HB + (size_t)(rbase + J * 16) * 128;
#pragma unroll
    for (int q = t; q < 512; q += 256) {
      int row = q >> 5;          // 32 float4 per row
      int c4  = (q & 31) * 4;
      async_copy_b128(&haI[row][c4], sAI + row * 128 + c4);
      async_copy_b128(&haJ[row][c4], sAJ + row * 128 + c4);
      async_copy_b128(&hbI[row][c4], sBI + row * 128 + c4);
      async_copy_b128(&hbJ[row][c4], sBJ + row * 128 + c4);
    }
    if (t < 32) async_copy_b128(&wv[t * 4], w + t * 4);
  }
  asm volatile("s_wait_asynccnt 0x0" ::: "memory");
  __syncthreads();

  const float bias = eb2[0];
  const int i = t >> 4;          // 0..15 within I-tile
  const int j = t & 15;          // 0..15 within J-tile

  // Vectorized LDS reads (ds_load_b128): 5 b128 loads + 24 VALU per 4 h.
  float x = 0.0f, y = 0.0f;
#pragma unroll 4
  for (int h = 0; h < 128; h += 4) {
    float4 ai = *(const float4*)&haI[i][h];
    float4 bi = *(const float4*)&hbI[i][h];
    float4 aj = *(const float4*)&haJ[j][h];
    float4 bj = *(const float4*)&hbJ[j][h];
    float4 wh = *(const float4*)&wv[h];
    float u0 = ai.x + bj.x, u1 = ai.y + bj.y, u2 = ai.z + bj.z, u3 = ai.w + bj.w;
    float v0 = aj.x + bi.x, v1 = aj.y + bi.y, v2 = aj.z + bi.z, v3 = aj.w + bi.w;
    u0 = u0 > 0.0f ? u0 : 0.0f;  u1 = u1 > 0.0f ? u1 : 0.0f;
    u2 = u2 > 0.0f ? u2 : 0.0f;  u3 = u3 > 0.0f ? u3 : 0.0f;
    v0 = v0 > 0.0f ? v0 : 0.0f;  v1 = v1 > 0.0f ? v1 : 0.0f;
    v2 = v2 > 0.0f ? v2 : 0.0f;  v3 = v3 > 0.0f ? v3 : 0.0f;
    x = fmaf(u0, wh.x, x); x = fmaf(u1, wh.y, x);
    x = fmaf(u2, wh.z, x); x = fmaf(u3, wh.w, x);
    y = fmaf(v0, wh.x, y); y = fmaf(v1, wh.y, y);
    y = fmaf(v2, wh.z, y); y = fmaf(v3, wh.w, y);
  }
  x += bias; y += bias;
  float px = 1.0f / (1.0f + expf(-x));
  float py = 1.0f / (1.0f + expf(-y));
  float val = 0.5f * (px + py);

  const size_t bb = (size_t)b * 1024 * 1024;
  const int gi = I * 16 + i, gj = J * 16 + j;
  __builtin_nontemporal_store(val, &adj[bb + (size_t)gi * 1024 + gj]);
  __builtin_nontemporal_store(val, &adj[bb + (size_t)gj * 1024 + gi]);
}

extern "C" void kernel_launch(void* const* d_in, const int* in_sizes, int n_in,
                              void* d_out, int out_size, void* d_ws, size_t ws_size,
                              hipStream_t stream) {
  const float* Z    = (const float*)d_in[0];  // node_z (2,1024,64)
  const float* W1   = (const float*)d_in[1];  // nW1 (64,128)
  const float* b1v  = (const float*)d_in[2];  // nb1 (128)
  const float* W2   = (const float*)d_in[3];  // nW2 (128,128)
  const float* b2v  = (const float*)d_in[4];  // nb2 (128)
  const float* eW1  = (const float*)d_in[5];  // eW1 (256,128)
  const float* eb1  = (const float*)d_in[6];  // eb1 (128)
  const float* eW2  = (const float*)d_in[7];  // eW2 (128,1)
  const float* eb2  = (const float*)d_in[8];  // eb2 (1)
  (void)in_sizes; (void)n_in; (void)out_size; (void)ws_size;

  float* out   = (float*)d_out;
  float* feats = out;                              // 2*1024*128 = 262144 f32
  float* adj   = out + (size_t)NROWS * NF;         // 2*1024*1024 f32

  float* HA = (float*)d_ws;                        // 2048*128 f32 = 1 MB
  float* HB = HA + (size_t)NROWS * HID;            // 1 MB

  gen_feats_kernel<<<NROWS / 16, 256, 0, stream>>>(Z, W1, b1v, W2, b2v, eW1, eb1,
                                                   feats, HA, HB);
  edge_kernel<<<dim3(64, 64, 2), 256, 0, stream>>>(HA, HB, eW2, eb2, adj);
}
